// PIAdapter_Attn_38405597561109
// MI455X (gfx1250) — matrix operand order repaired and checked by reading, not compiled, not run on hardware
//
#include <hip/hip_runtime.h>
#include <hip/hip_bf16.h>

typedef __attribute__((ext_vector_type(16))) _Float16 v16h;
typedef __attribute__((ext_vector_type(8)))  float    v8f;

namespace {
constexpr int kB  = 16;
constexpr int kL  = 4096;
constexpr int kD  = 768;
constexpr int kH  = 8;
constexpr int kDG = 384;          // D / G
constexpr int kHG = 4;            // H / G
constexpr int kKSteps = kD / 32;  // 24 WMMA K-steps
constexpr int kTokPerBlk = 4;     // out_kernel tokens per block
constexpr float kInvT = 0.1f;     // 1 / T

// workspace byte offsets (all 32B-aligned)
constexpr size_t kOffCRaw   = 0;                                   // [B,L,8] f32   = 2,097,152 B
constexpr size_t kOffPooled = kOffCRaw   + (size_t)kB*kL*kH*4;     // [B,768] f32   =    49,152 B
constexpr size_t kOffCache  = kOffPooled + (size_t)kB*kD*4;        // [B,8]   f32   =       512 B
constexpr size_t kOffNmask  = kOffCache  + (size_t)kB*kH*4;        // [B]     f32   =        64 B
constexpr size_t kOffWGate  = kOffNmask  + (size_t)kB*4;           // [B,2]   f32   =       128 B
constexpr size_t kOffCacheF = kOffWGate  + (size_t)kB*2*4;         // [B,8]   f32   =       512 B
constexpr size_t kOffWFrag  = kOffCacheF + (size_t)kB*kH*4;        // 24*32*16 f16  =    24,576 B
constexpr size_t kZeroBytes = kOffWGate - kOffPooled;              // pooled+cache+nmask
} // namespace

// ---------------------------------------------------------------------------
// Pre-pack WA (cols 0-7) and WC (cols 8-15) into this file's contiguous-K B-fragment order:
// 16-bit B 32x16, lanes 0-15 hold K=kb+0..15, lanes 16-31 hold K=kb+16..31,
// column N = lane & 15.  One v16h (32B) per lane per K-step.
// ---------------------------------------------------------------------------
__global__ void prep_wfrag_kernel(const float* __restrict__ WA,
                                  const float* __restrict__ WC,
                                  _Float16* __restrict__ wfrag) {
    const int s   = blockIdx.x;       // K-step 0..23
    const int n   = threadIdx.x;      // lane 0..31
    const int col = n & 15;
    const int kb  = s * 32 + ((n >> 4) << 4);
    const float* src = (col < kH) ? (WA + (size_t)col * kD)
                                  : (WC + (size_t)(col - kH) * kD);
    _Float16* dst = wfrag + ((size_t)(s * 32 + n)) * 16;
#pragma unroll
    for (int e = 0; e < 16; ++e) dst[e] = (_Float16)src[kb + e];
}

// ---------------------------------------------------------------------------
// Pass 1: single sweep over x.
//  - WMMA: [16 tok x 768] * [768 x 16]  -> cols 0-7 = x*WA^T, cols 8-15 = x*WC^T
//  - masked reduce of the WA half into cache[b,h]
//  - store WC half (raw c) to workspace
//  - masked column-sum pooled[b,d] (register accum, tile is cache-hot)
// grid = (L/128, B), block = 256 (8 waves, 16 tokens each)
// ---------------------------------------------------------------------------
__global__ void __launch_bounds__(256)
proj_kernel(const float* __restrict__ x, const int* __restrict__ qmask,
            const _Float16* __restrict__ wfrag,
            float* __restrict__ cRaw, float* __restrict__ pooledAcc,
            float* __restrict__ cacheAcc, float* __restrict__ nmaskAcc) {
    const int b       = blockIdx.y;
    const int tid     = threadIdx.x;
    const int wave    = tid >> 5;
    const int lane    = tid & 31;
    const int tokBase = blockIdx.x * 128;
    const int waveTok = tokBase + wave * 16;

    __shared__ float maskLDS[128];
    __shared__ float cacheLDS[kH];
    if (tid < kH) cacheLDS[tid] = 0.0f;
    if (tid < 128)
        maskLDS[tid] = (float)qmask[(size_t)b * kL + tokBase + tid];
    __syncthreads();

    // ---- pooled[b, d] masked column sum: thread owns d = tid, tid+256, tid+512
    {
        float a0 = 0.f, a1 = 0.f, a2 = 0.f;
        for (int tt = 0; tt < 128; ++tt) {
            const float m = maskLDS[tt];
            const float* xr = x + ((size_t)b * kL + tokBase + tt) * kD;
            a0 += m * xr[tid];
            a1 += m * xr[tid + 256];
            a2 += m * xr[tid + 512];
        }
        atomicAdd(&pooledAcc[b * kD + tid      ], a0);
        atomicAdd(&pooledAcc[b * kD + tid + 256], a1);
        atomicAdd(&pooledAcc[b * kD + tid + 512], a2);
    }

    // ---- WMMA chain: D(16x16) += A(16x32) x B(32x16), 24 K-steps
    const int t     = waveTok + (lane & 15);     // A-matrix row (token)
    const int koffA = (lane >> 4) << 4;          // 0 or 16 (contiguous K, matches wfrag)
    v8f acc = {};
    for (int s = 0; s < kKSteps; ++s) {
        const float* xa = x + ((size_t)b * kL + t) * kD + s * 32 + koffA;
        const float4 f0 = *reinterpret_cast<const float4*>(xa);
        const float4 f1 = *reinterpret_cast<const float4*>(xa + 4);
        const float4 f2 = *reinterpret_cast<const float4*>(xa + 8);
        const float4 f3 = *reinterpret_cast<const float4*>(xa + 12);
        v16h a;
        a[0]  = (_Float16)f0.x; a[1]  = (_Float16)f0.y;
        a[2]  = (_Float16)f0.z; a[3]  = (_Float16)f0.w;
        a[4]  = (_Float16)f1.x; a[5]  = (_Float16)f1.y;
        a[6]  = (_Float16)f1.z; a[7]  = (_Float16)f1.w;
        a[8]  = (_Float16)f2.x; a[9]  = (_Float16)f2.y;
        a[10] = (_Float16)f2.z; a[11] = (_Float16)f2.w;
        a[12] = (_Float16)f3.x; a[13] = (_Float16)f3.y;
        a[14] = (_Float16)f3.z; a[15] = (_Float16)f3.w;
        const v16h bf = *reinterpret_cast<const v16h*>(wfrag + ((size_t)(s * 32 + lane)) * 16);
        acc = __builtin_amdgcn_wmma_f32_16x16x32_f16(
            /*neg_a=*/false, a, /*neg_b=*/false, bf,
            /*c_mod=*/(short)0, acc, /*reuse_a=*/false, /*reuse_b=*/false);
    }

    // ---- epilogue: single lane-group divergence (not per element).
    // C/D layout: VGPR r -> row = r + 8*(lane>=16), col = lane & 15.
    const int col    = lane & 15;
    const int rowOff = (lane >> 4) << 3;
    if (col < kH) {
        // masked a-sum into cache[b, col]
        float myCache = 0.0f;
#pragma unroll
        for (int r = 0; r < 8; ++r)
            myCache += acc[r] * maskLDS[waveTok - tokBase + rowOff + r];
        atomicAdd(&cacheLDS[col], myCache);
    } else {
        // raw c store: for fixed r, 8 lanes write 8 consecutive floats (32B)
        float* cdst = cRaw + ((size_t)b * kL + waveTok + rowOff) * kH + (col - kH);
#pragma unroll
        for (int r = 0; r < 8; ++r) cdst[(size_t)r * kH] = acc[r];
    }
    __syncthreads();
    if (tid < kH) atomicAdd(&cacheAcc[b * kH + tid], cacheLDS[tid]);
    if (tid == 0) {
        float nm = 0.f;
        for (int i = 0; i < 128; ++i) nm += maskLDS[i];
        atomicAdd(&nmaskAcc[b], nm);
    }
}

// ---------------------------------------------------------------------------
// Gate: logits = pooled @ We^T + be ; w = softmax(logits / T)
// Also finalize cacheFull = cacheAcc + nmask * bA.  grid = (B), block = 256.
// ---------------------------------------------------------------------------
__global__ void __launch_bounds__(256)
gate_kernel(const float* __restrict__ pooledAcc, const float* __restrict__ We,
            const float* __restrict__ be, const float* __restrict__ cacheAcc,
            const float* __restrict__ nmaskAcc, const float* __restrict__ bA,
            float* __restrict__ wGate, float* __restrict__ cacheFull) {
    const int b = blockIdx.x;
    const int tid = threadIdx.x;
    __shared__ float r0[256], r1[256];
    float p0 = 0.f, p1 = 0.f;
    for (int d = tid; d < kD; d += 256) {
        const float pv = pooledAcc[b * kD + d];
        p0 += pv * We[d];
        p1 += pv * We[kD + d];
    }
    r0[tid] = p0; r1[tid] = p1;
    __syncthreads();
    for (int s = 128; s > 0; s >>= 1) {
        if (tid < s) { r0[tid] += r0[tid + s]; r1[tid] += r1[tid + s]; }
        __syncthreads();
    }
    if (tid == 0) {
        const float l0 = (r0[0] + be[0]) * kInvT;
        const float l1 = (r1[0] + be[1]) * kInvT;
        const float mx = fmaxf(l0, l1);
        const float e0 = __expf(l0 - mx), e1 = __expf(l1 - mx);
        const float inv = 1.0f / (e0 + e1);
        wGate[b * 2 + 0] = e0 * inv;
        wGate[b * 2 + 1] = e1 * inv;
    }
    if (tid < kH)
        cacheFull[b * kH + tid] = cacheAcc[b * kH + tid] + nmaskAcc[b] * bA[tid];
}

// ---------------------------------------------------------------------------
// Pass 2: per token, h = 0.5*(cacheFull + cRaw + bC) (8 floats), then grouped
// 4-wide up-projections + gate mix + residual.
// grid = (L/4, B), block = 256; 4 tokens per block.
// ---------------------------------------------------------------------------
__global__ void __launch_bounds__(256)
out_kernel(const float* __restrict__ x, const float* __restrict__ cRaw,
           const float* __restrict__ cacheFull, const float* __restrict__ bC,
           const float* __restrict__ WB, const float* __restrict__ bB,
           const float* __restrict__ WD, const float* __restrict__ bD,
           const float* __restrict__ wGate, float* __restrict__ out) {
    const int b = blockIdx.y, tid = threadIdx.x;
    const int l0 = blockIdx.x * kTokPerBlk;
    __shared__ float hsh[kTokPerBlk][kH];
    __shared__ float ws0, ws1;
    if (tid < kTokPerBlk * kH) {
        const int tt = tid / kH, hh = tid % kH;
        hsh[tt][hh] = 0.5f * (cacheFull[b * kH + hh] +
                              cRaw[((size_t)b * kL + l0 + tt) * kH + hh] + bC[hh]);
    }
    if (tid == 0) { ws0 = wGate[b * 2]; ws1 = wGate[b * 2 + 1]; }
    __syncthreads();
    const float w0 = ws0, w1 = ws1;
    for (int d = tid; d < kD; d += 256) {
        const int gOff = (d >= kDG) ? kHG : 0;
        float wb0 = WB[d * kHG + 0], wb1 = WB[d * kHG + 1];
        float wb2 = WB[d * kHG + 2], wb3 = WB[d * kHG + 3];
        float wd0 = WD[d * kHG + 0], wd1 = WD[d * kHG + 1];
        float wd2 = WD[d * kHG + 2], wd3 = WD[d * kHG + 3];
        const float bBd = bB[d], bDd = bD[d];
#pragma unroll
        for (int tt = 0; tt < kTokPerBlk; ++tt) {
            const float* hp = &hsh[tt][gOff];
            const float yB = bBd + wb0 * hp[0] + wb1 * hp[1] + wb2 * hp[2] + wb3 * hp[3];
            const float yD = bDd + wd0 * hp[0] + wd1 * hp[1] + wd2 * hp[2] + wd3 * hp[3];
            const size_t idx = ((size_t)b * kL + l0 + tt) * kD + d;
            out[idx] = x[idx] + w0 * yB + w1 * yD;
        }
    }
}

// ---------------------------------------------------------------------------
extern "C" void kernel_launch(void* const* d_in, const int* in_sizes, int n_in,
                              void* d_out, int out_size, void* d_ws, size_t ws_size,
                              hipStream_t stream) {
    (void)in_sizes; (void)n_in; (void)out_size; (void)ws_size;
    const float* x     = (const float*)d_in[0];
    const int*   qmask = (const int*)  d_in[1];
    const float* WA    = (const float*)d_in[2];
    const float* bA    = (const float*)d_in[3];
    const float* WC    = (const float*)d_in[4];
    const float* bC    = (const float*)d_in[5];
    const float* WB    = (const float*)d_in[6];
    const float* bB    = (const float*)d_in[7];
    const float* WD    = (const float*)d_in[8];
    const float* bD    = (const float*)d_in[9];
    const float* We    = (const float*)d_in[10];
    const float* be    = (const float*)d_in[11];
    float* out = (float*)d_out;

    char* ws = (char*)d_ws;
    float*    cRaw      = (float*)   (ws + kOffCRaw);
    float*    pooledAcc = (float*)   (ws + kOffPooled);
    float*    cacheAcc  = (float*)   (ws + kOffCache);
    float*    nmaskAcc  = (float*)   (ws + kOffNmask);
    float*    wGate     = (float*)   (ws + kOffWGate);
    float*    cacheFull = (float*)   (ws + kOffCacheF);
    _Float16* wfrag     = (_Float16*)(ws + kOffWFrag);

    // zero the accumulators (graph-capture safe)
    hipMemsetAsync(ws + kOffPooled, 0, kZeroBytes, stream);

    prep_wfrag_kernel<<<dim3(kKSteps), dim3(32), 0, stream>>>(WA, WC, wfrag);

    proj_kernel<<<dim3(kL / 128, kB), dim3(256), 0, stream>>>(
        x, qmask, wfrag, cRaw, pooledAcc, cacheAcc, nmaskAcc);

    gate_kernel<<<dim3(kB), dim3(256), 0, stream>>>(
        pooledAcc, We, be, cacheAcc, nmaskAcc, bA, wGate, cacheFull);

    out_kernel<<<dim3(kL / kTokPerBlk, kB), dim3(256), 0, stream>>>(
        x, cRaw, cacheFull, bC, WB, bB, WD, bD, wGate, out);
}
